// QGNet_75642964017649
// MI455X (gfx1250) — matrix operand order repaired and checked by reading, not compiled, not run on hardware
//
#include <hip/hip_runtime.h>
#include <cstdint>
#include <cstddef>

typedef __bf16 bf16_t;
typedef __attribute__((ext_vector_type(16))) __bf16 v16bf;
typedef __attribute__((ext_vector_type(8)))  __bf16 v8bf;
typedef __attribute__((ext_vector_type(8)))  float  v8f;

#define NV_  32000
#define NS_  400
#define NB_  32
#define NT_  30
#define NE_  300
#define NH_  600
#define SB_  12800   // S*B
// padded K dims (multiples of 32); N dims multiples of 32 (H -> 608)
#define KP_IN0 352
#define KP_E   320
#define KP_H   608
#define KP_2H  1216
#define KP_3H  1824
#define N4H    2400

__device__ __forceinline__ float sigmf(float x) { return 1.f / (1.f + __expf(-x)); }
__device__ __forceinline__ float seluf(float x) {
  const float l = 1.0507009873554805f, a = 1.6732632423543772f;
  return x > 0.f ? l * x : l * a * (__expf(x) - 1.f);
}

// Build a 16x32 A fragment (ISA layout) from two aligned 16B loads.
__device__ __forceinline__ v16bf load_a_frag(const bf16_t* arow /* row base + k0 */, int g) {
  v8bf lo = *(const v8bf*)(arow + 8 * g);
  v8bf hi = *(const v8bf*)(arow + 16 + 8 * g);
  return __builtin_shufflevector(lo, hi, 0, 1, 2, 3, 4, 5, 6, 7,
                                 8, 9, 10, 11, 12, 13, 14, 15);
}

// ---------------- generic WMMA bf16 GEMM: C = (accum?C:0) + A@Bp (+bias)(+act) ------
// A: M x K (lda) bf16 row-major.  Bp: packed fragment-order B, blocks of 512 bf16 per
// (32k x 16n) tile: offset = ((k0/32)*ntiles + n0/16)*512 + lane*16 + e.
// Each wave computes a 16x32 C strip (2 N-tiles share one A fragment).
__global__ __launch_bounds__(256) void gemm_bf16_kernel(
    const bf16_t* __restrict__ A, int lda,
    const bf16_t* __restrict__ Bp,
    float* __restrict__ C, int ldc,
    int M, int N, int K,
    const float* __restrict__ bias, int act, int accum) {
  const int gw   = (blockIdx.x * blockDim.x + threadIdx.x) >> 5;
  const int lane = threadIdx.x & 31;
  const int ntp   = N >> 5;                 // N-tile pairs
  const int tiles = (M >> 4) * ntp;
  if (gw >= tiles) return;                  // wave-uniform exit: EXEC stays all-1s
  const int m0 = (gw / ntp) << 4;
  const int n0 = (gw % ntp) << 5;
  const int g  = lane >> 4;
  const int n  = n0 + (lane & 15);
  const int m  = m0 + (lane & 15);
  v8f acc0, acc1;
  if (accum) {
#pragma unroll
    for (int r = 0; r < 8; ++r) {
      acc0[r] = C[(size_t)(m0 + r + (g << 3)) * ldc + n];
      acc1[r] = C[(size_t)(m0 + r + (g << 3)) * ldc + n + 16];
    }
  } else {
#pragma unroll
    for (int r = 0; r < 8; ++r) { acc0[r] = 0.f; acc1[r] = 0.f; }
  }
  const bf16_t* arow = A + (size_t)m * lda;
  const bf16_t* bp   = Bp + (((size_t)(n0 >> 4)) << 9) + (lane << 4);
  const size_t  bstep = (size_t)(N >> 4) << 9;   // next k-block
  for (int k0 = 0; k0 < K; k0 += 32, bp += bstep) {
    v16bf fa  = load_a_frag(arow + k0, g);       // 2x b128
    v16bf fb0 = *(const v16bf*)bp;               // 2x b128
    v16bf fb1 = *(const v16bf*)(bp + 512);       // 2x b128 (const +1024B offset)
    if (k0 + 32 < K) __builtin_prefetch(bp + bstep, 0, 1);  // global_prefetch_b8
    acc0 = __builtin_amdgcn_wmma_f32_16x16x32_bf16(false, fa, false, fb0, (short)0, acc0,
                                                   false, false);
    acc1 = __builtin_amdgcn_wmma_f32_16x16x32_bf16(false, fa, false, fb1, (short)0, acc1,
                                                   false, false);
  }
  const float bv0 = bias ? bias[n] : 0.f;
  const float bv1 = bias ? bias[n + 16] : 0.f;
#pragma unroll
  for (int r = 0; r < 8; ++r) {
    float x0 = acc0[r] + bv0, x1 = acc1[r] + bv1;
    if (act == 1) { x0 = seluf(x0); x1 = seluf(x1); }
    C[(size_t)(m0 + r + (g << 3)) * ldc + n]      = x0;
    C[(size_t)(m0 + r + (g << 3)) * ldc + n + 16] = x1;
  }
}

// ---------------- fused bidirectional LSTM scan (one WG per direction) --------------
struct ScanDir {
  const float*  Xi;     // (S*B, 2400) precomputed x@Wih
  const bf16_t* Whh;    // packed-B (608 x 2400)
  const float*  b;      // (2400)
  float*        Gbuf;   // (32, 2400) scratch for h@Whh
  bf16_t*       out_bf; // (S*B, out_ld), writes cols [coloff, coloff+600)
  float*        out_f32;// optional (S*B, 1200); nullptr if unused
  int reverse;
  int coloff;
};

__global__ __launch_bounds__(1024) void lstm_scan_kernel(ScanDir da, ScanDir db, int out_ld) {
  const ScanDir d = (blockIdx.x == 0) ? da : db;
  __shared__ bf16_t hS[NB_ * KP_H];     // 38.9 KB
  const int tid  = threadIdx.x;
  const int wave = tid >> 5, lane = tid & 31;
  for (int i = tid; i < NB_ * KP_H; i += 1024) hS[i] = (bf16_t)0.f;
  float cloc[19];
#pragma unroll
  for (int q = 0; q < 19; ++q) cloc[q] = 0.f;
  __syncthreads();

  const int g = lane >> 4;
  for (int step = 0; step < NS_; ++step) {
    const int s = d.reverse ? (NS_ - 1 - step) : step;
    // ---- Gbuf(32,2400) = hS(32x608,bf16) @ Whh(608x2400), 16x32 strip per wave ----
    for (int tp = wave; tp < 2 * 75; tp += 32) {   // wave-uniform loop
      const int m0 = (tp / 75) << 4;
      const int n0 = (tp % 75) << 5;
      const int n  = n0 + (lane & 15);
      const int m  = m0 + (lane & 15);
      v8f acc0, acc1;
#pragma unroll
      for (int r = 0; r < 8; ++r) { acc0[r] = 0.f; acc1[r] = 0.f; }
      const bf16_t* arow = hS + m * KP_H;
      const bf16_t* bp   = d.Whh + (((size_t)(n0 >> 4)) << 9) + (lane << 4);
      const size_t  bstep = (size_t)150 << 9;
      for (int k0 = 0; k0 < KP_H; k0 += 32, bp += bstep) {
        v16bf fa  = load_a_frag(arow + k0, g);   // ds_load_b128 x2
        v16bf fb0 = *(const v16bf*)bp;
        v16bf fb1 = *(const v16bf*)(bp + 512);
        acc0 = __builtin_amdgcn_wmma_f32_16x16x32_bf16(false, fa, false, fb0, (short)0, acc0,
                                                       false, false);
        acc1 = __builtin_amdgcn_wmma_f32_16x16x32_bf16(false, fa, false, fb1, (short)0, acc1,
                                                       false, false);
      }
#pragma unroll
      for (int r = 0; r < 8; ++r) {
        d.Gbuf[(m0 + r + (g << 3)) * N4H + n]      = acc0[r];
        d.Gbuf[(m0 + r + (g << 3)) * N4H + n + 16] = acc1[r];
      }
    }
    __threadfence();
    __syncthreads();
    // ---- gates: c,h update; h -> LDS (bf16) + global outputs ----
    int q = 0;
    for (int i = tid; i < NB_ * NH_; i += 1024, ++q) {
      const int bb = i / NH_, j = i % NH_;
      const float* xr = d.Xi + (size_t)(s * NB_ + bb) * N4H;
      const float* gr = d.Gbuf + bb * N4H;
      const float gi = xr[j]        + gr[j]        + d.b[j];
      const float gf = xr[j + 600]  + gr[j + 600]  + d.b[j + 600];
      const float gg = xr[j + 1200] + gr[j + 1200] + d.b[j + 1200];
      const float go = xr[j + 1800] + gr[j + 1800] + d.b[j + 1800];
      const float c  = sigmf(gf) * cloc[q] + sigmf(gi) * tanhf(gg);
      const float h  = sigmf(go) * tanhf(c);
      cloc[q] = c;
      hS[bb * KP_H + j] = (bf16_t)h;
      const size_t row = (size_t)(s * NB_ + bb);
      d.out_bf[row * out_ld + d.coloff + j] = (bf16_t)h;
      if (d.out_f32) d.out_f32[row * (2 * NH_) + d.coloff + j] = h;
    }
    __threadfence();
    __syncthreads();
  }
}

// ---------------- helper kernels ----------------------------------------------------
__global__ void zero_kernel(uint32_t* p, size_t n32) {
  size_t i = blockIdx.x * (size_t)blockDim.x + threadIdx.x;
  for (; i < n32; i += (size_t)gridDim.x * blockDim.x) p[i] = 0u;
}

// f32 (rows x cols) weight -> bf16 packed fragment-order B (Kp x N), zero padded.
__global__ void pack_b_kernel(const float* __restrict__ src, int rows, int cols,
                              bf16_t* __restrict__ dst, int Kp, int N) {
  size_t i = blockIdx.x * (size_t)blockDim.x + threadIdx.x;
  const size_t tot = (size_t)(Kp >> 5) * (N >> 4) * 512;
  if (i >= tot) return;
  const int blk = (int)(i >> 9);
  const int r   = (int)(i & 511);
  const int lane = r >> 4, e = r & 15;
  const int ntiles = N >> 4;
  const int kb = blk / ntiles, nb = blk % ntiles;
  const int k = (kb << 5) + ((lane >> 4) << 3) + (e < 8 ? e : 8 + e);
  const int n = (nb << 4) + (lane & 15);
  const float v = (k < rows && n < cols) ? src[(size_t)k * cols + n] : 0.f;
  dst[i] = (bf16_t)v;
}

__global__ void embed_concat_kernel(const int* __restrict__ src,
                                    const int* __restrict__ f0, const int* __restrict__ f1,
                                    const int* __restrict__ f2, const int* __restrict__ f3,
                                    const float* __restrict__ emb,
                                    const float* __restrict__ fe0, const float* __restrict__ fe1,
                                    const float* __restrict__ fe2, const float* __restrict__ fe3,
                                    bf16_t* __restrict__ X) {
  size_t i = blockIdx.x * (size_t)blockDim.x + threadIdx.x;
  if (i >= (size_t)SB_ * KP_IN0) return;
  const int row = (int)(i / KP_IN0), c = (int)(i % KP_IN0);
  float v = 0.f;
  if      (c < 300) v = emb[(size_t)src[row] * NE_ + c];
  else if (c < 314) v = fe0[f0[row] * 14 + (c - 300)];
  else if (c < 321) v = fe1[f1[row] * 7  + (c - 314)];
  else if (c < 326) v = fe2[f2[row] * 5  + (c - 321)];
  else if (c < 336) v = fe3[f3[row] * 10 + (c - 326)];
  X[i] = (bf16_t)v;
}

__global__ void dec_embed_kernel(const int* __restrict__ trg, int t,
                                 const float* __restrict__ emb, bf16_t* __restrict__ ebuf) {
  int i = blockIdx.x * blockDim.x + threadIdx.x;
  if (i >= NB_ * KP_E) return;
  const int bb = i / KP_E, j = i % KP_E;
  const int tok = trg[t * NB_ + bb];
  ebuf[i] = (bf16_t)((j < NE_) ? emb[(size_t)tok * NE_ + j] : 0.f);
}

__global__ void dec_gate_kernel(const float* __restrict__ G, const float* __restrict__ bias,
                                float* __restrict__ c, float* __restrict__ hf,
                                bf16_t* __restrict__ hbf) {
  int i = blockIdx.x * blockDim.x + threadIdx.x;
  if (i >= NB_ * NH_) return;
  const int bb = i / NH_, j = i % NH_;
  const float* g = G + bb * N4H;
  const float gi = g[j] + bias[j], gf = g[j + 600] + bias[j + 600];
  const float gg = g[j + 1200] + bias[j + 1200], go = g[j + 1800] + bias[j + 1800];
  const float cc = sigmf(gf) * c[i] + sigmf(gi) * tanhf(gg);
  const float hh = sigmf(go) * tanhf(cc);
  c[i] = cc; hf[i] = hh; hbf[bb * KP_H + j] = (bf16_t)hh;
}

// scores[b*S+s] = sum_h tanh(x1[s*B+b,h] + q[b,h] + b_attn[h]) * v[h]; wave per (s,b)
__global__ __launch_bounds__(256) void attn_score_kernel(
    const float* __restrict__ x1, const float* __restrict__ q,
    const float* __restrict__ b_attn, const float* __restrict__ v,
    float* __restrict__ scores) {
  const int gw = (blockIdx.x * blockDim.x + threadIdx.x) >> 5;
  const int lane = threadIdx.x & 31;
  if (gw >= SB_) return;
  const int s = gw / NB_, bb = gw % NB_;
  const float* xr = x1 + (size_t)gw * KP_H;
  const float* qr = q + bb * KP_H;
  float acc = 0.f;
  for (int h = lane; h < NH_; h += 32) acc += tanhf(xr[h] + qr[h] + b_attn[h]) * v[h];
#pragma unroll
  for (int off = 16; off; off >>= 1) acc += __shfl_down(acc, off, 32);
  if (lane == 0) scores[bb * NS_ + s] = acc;
}

__global__ __launch_bounds__(256) void softmax_kernel(const float* __restrict__ scores,
                                                      float* __restrict__ attn) {
  const int bb = blockIdx.x;
  __shared__ float red[256];
  const float* row = scores + bb * NS_;
  float m = -1e30f;
  for (int s = threadIdx.x; s < NS_; s += 256) m = fmaxf(m, row[s]);
  red[threadIdx.x] = m; __syncthreads();
  for (int w = 128; w; w >>= 1) {
    if (threadIdx.x < w) red[threadIdx.x] = fmaxf(red[threadIdx.x], red[threadIdx.x + w]);
    __syncthreads();
  }
  m = red[0]; __syncthreads();
  float sum = 0.f;
  for (int s = threadIdx.x; s < NS_; s += 256) sum += __expf(row[s] - m);
  red[threadIdx.x] = sum; __syncthreads();
  for (int w = 128; w; w >>= 1) {
    if (threadIdx.x < w) red[threadIdx.x] += red[threadIdx.x + w];
    __syncthreads();
  }
  const float inv = 1.f / red[0];
  for (int s = threadIdx.x; s < NS_; s += 256) attn[bb * NS_ + s] = __expf(row[s] - m) * inv;
}

__global__ void context_kernel(const float* __restrict__ attn, const float* __restrict__ eo,
                               float* __restrict__ ctx) {
  int i = blockIdx.x * blockDim.x + threadIdx.x;
  if (i >= NB_ * 2 * NH_) return;
  const int bb = i / (2 * NH_), dd = i % (2 * NH_);
  float acc = 0.f;
  for (int s = 0; s < NS_; ++s)
    acc += attn[bb * NS_ + s] * eo[(size_t)(s * NB_ + bb) * (2 * NH_) + dd];
  ctx[i] = acc;
}

__global__ void build_gs_kernel(const float* __restrict__ h1f, const float* __restrict__ ctx,
                                bf16_t* __restrict__ gs) {
  int i = blockIdx.x * blockDim.x + threadIdx.x;
  if (i >= NB_ * KP_3H) return;
  const int bb = i / KP_3H, j = i % KP_3H;
  float v = 0.f;
  if (j < NH_) v = h1f[bb * NH_ + j];
  else if (j < 3 * NH_) v = ctx[bb * (2 * NH_) + (j - NH_)];
  gs[i] = (bf16_t)v;
}

__global__ void pz_kernel(const float* __restrict__ h1f, const float* __restrict__ wz,
                          const float* __restrict__ bz, float* __restrict__ pz) {
  const int bb = threadIdx.x;
  if (bb >= NB_) return;
  float acc = bz[0];
  for (int j = 0; j < NH_; ++j) acc += h1f[bb * NH_ + j] * wz[j];
  pz[bb] = sigmf(acc);
}

__global__ void scale_store_kernel(const float* __restrict__ pv, const float* __restrict__ pz,
                                   float* __restrict__ out, int t) {
  size_t i = blockIdx.x * (size_t)blockDim.x + threadIdx.x;
  if (i >= (size_t)NB_ * NV_) return;
  const int bb = (int)(i / NV_);
  out[(size_t)t * NB_ * NV_ + i] = pz[bb] * pv[i];
}

__global__ void scatter_kernel(const int* __restrict__ src, const float* __restrict__ attn,
                               const float* __restrict__ pz, float* __restrict__ out, int t) {
  int i = blockIdx.x * blockDim.x + threadIdx.x;
  if (i >= NB_ * NS_) return;
  const int bb = i / NS_, s = i % NS_;
  const int tok = src[s * NB_ + bb];
  atomicAdd(out + (size_t)t * NB_ * NV_ + (size_t)bb * NV_ + tok,
            (1.f - pz[bb]) * attn[i]);
}

// =====================================================================================
extern "C" void kernel_launch(void* const* d_in, const int* in_sizes, int n_in,
                              void* d_out, int out_size, void* d_ws, size_t ws_size,
                              hipStream_t stream) {
  (void)in_sizes; (void)n_in; (void)out_size; (void)ws_size;
  // inputs (setup_inputs order; params pytree-flattened with sorted dict keys)
  const int*   src    = (const int*)d_in[0];
  const int*   feat0  = (const int*)d_in[2];
  const int*   feat1  = (const int*)d_in[3];
  const int*   feat2  = (const int*)d_in[4];
  const int*   feat3  = (const int*)d_in[5];
  const int*   trg    = (const int*)d_in[7];
  const float* Vb     = (const float*)d_in[8];
  const float* Vw     = (const float*)d_in[9];   // (1800, 32000)
  const float* W_h    = (const float*)d_in[10];  // (1200, 600)
  const float* W_s    = (const float*)d_in[11];  // (600, 600)
  const float* b_attn = (const float*)d_in[12];
  const float* bz     = (const float*)d_in[13];
  const float* d0Whh  = (const float*)d_in[14];
  const float* d0Wih  = (const float*)d_in[15];
  const float* d0b    = (const float*)d_in[16];
  const float* d1Whh  = (const float*)d_in[17];
  const float* d1Wih  = (const float*)d_in[18];
  const float* d1b    = (const float*)d_in[19];
  const float* emb    = (const float*)d_in[20];  // (32000, 300)
  const float* l0bWhh = (const float*)d_in[21];
  const float* l0bWih = (const float*)d_in[22];
  const float* l0bb   = (const float*)d_in[23];
  const float* l0fWhh = (const float*)d_in[24];
  const float* l0fWih = (const float*)d_in[25];
  const float* l0fb   = (const float*)d_in[26];
  const float* l1bWhh = (const float*)d_in[27];
  const float* l1bWih = (const float*)d_in[28];
  const float* l1bb   = (const float*)d_in[29];
  const float* l1fWhh = (const float*)d_in[30];
  const float* l1fWih = (const float*)d_in[31];
  const float* l1fb   = (const float*)d_in[32];
  const float* fe0    = (const float*)d_in[33];
  const float* fe1    = (const float*)d_in[34];
  const float* fe2    = (const float*)d_in[35];
  const float* fe3    = (const float*)d_in[36];
  const float* v_vec  = (const float*)d_in[37];
  const float* wz     = (const float*)d_in[38];
  float* out = (float*)d_out;

  // workspace carve-up
  char* wsb = (char*)d_ws;
  size_t off = 0;
  auto alloc = [&](size_t bytes) -> char* {
    char* p = wsb + off;
    off = (off + bytes + 255) & ~(size_t)255;
    return p;
  };
  bf16_t* wih_l0f = (bf16_t*)alloc((size_t)KP_IN0 * N4H * 2);
  bf16_t* wih_l0b = (bf16_t*)alloc((size_t)KP_IN0 * N4H * 2);
  bf16_t* whh_l0f = (bf16_t*)alloc((size_t)KP_H * N4H * 2);
  bf16_t* whh_l0b = (bf16_t*)alloc((size_t)KP_H * N4H * 2);
  bf16_t* wih_l1f = (bf16_t*)alloc((size_t)KP_2H * N4H * 2);
  bf16_t* wih_l1b = (bf16_t*)alloc((size_t)KP_2H * N4H * 2);
  bf16_t* whh_l1f = (bf16_t*)alloc((size_t)KP_H * N4H * 2);
  bf16_t* whh_l1b = (bf16_t*)alloc((size_t)KP_H * N4H * 2);
  bf16_t* wih_d0  = (bf16_t*)alloc((size_t)KP_E * N4H * 2);
  bf16_t* whh_d0  = (bf16_t*)alloc((size_t)KP_H * N4H * 2);
  bf16_t* wih_d1  = (bf16_t*)alloc((size_t)KP_H * N4H * 2);
  bf16_t* whh_d1  = (bf16_t*)alloc((size_t)KP_H * N4H * 2);
  bf16_t* wh_bf   = (bf16_t*)alloc((size_t)KP_2H * KP_H * 2);
  bf16_t* wsm_bf  = (bf16_t*)alloc((size_t)KP_H * KP_H * 2);
  bf16_t* vw_bf   = (bf16_t*)alloc((size_t)KP_3H * NV_ * 2);
  bf16_t* xbf     = (bf16_t*)alloc((size_t)SB_ * KP_IN0 * 2);
  bf16_t* l0bf    = (bf16_t*)alloc((size_t)SB_ * KP_2H * 2);
  bf16_t* eobf    = (bf16_t*)alloc((size_t)SB_ * KP_2H * 2);
  float*  xiA     = (float*)alloc((size_t)SB_ * N4H * 4);
  float*  xiB     = (float*)alloc((size_t)SB_ * N4H * 4);
  float*  eof     = (float*)alloc((size_t)SB_ * 2 * NH_ * 4);
  float*  x1      = (float*)alloc((size_t)SB_ * KP_H * 4);
  float*  gbuf    = (float*)alloc((size_t)2 * NB_ * N4H * 4);
  float*  gd      = (float*)alloc((size_t)NB_ * N4H * 4);
  float*  qbuf    = (float*)alloc((size_t)NB_ * KP_H * 4);
  float*  scores  = (float*)alloc((size_t)NB_ * NS_ * 4);
  float*  attn    = (float*)alloc((size_t)NB_ * NS_ * 4);
  float*  ctx     = (float*)alloc((size_t)NB_ * 2 * NH_ * 4);
  float*  pv      = (float*)alloc((size_t)NB_ * NV_ * 4);
  float*  h0f     = (float*)alloc((size_t)NB_ * NH_ * 4);
  float*  c0f     = (float*)alloc((size_t)NB_ * NH_ * 4);
  float*  h1f     = (float*)alloc((size_t)NB_ * NH_ * 4);
  float*  c1f     = (float*)alloc((size_t)NB_ * NH_ * 4);
  float*  pz      = (float*)alloc((size_t)NB_ * 4);
  bf16_t* h0bf    = (bf16_t*)alloc((size_t)NB_ * KP_H * 2);
  bf16_t* h1bf    = (bf16_t*)alloc((size_t)NB_ * KP_H * 2);
  bf16_t* ebuf    = (bf16_t*)alloc((size_t)NB_ * KP_E * 2);
  bf16_t* gsbf    = (bf16_t*)alloc((size_t)NB_ * KP_3H * 2);

  auto zero = [&](void* p, size_t bytes) {
    size_t n32 = bytes >> 2;
    int blk = (int)((n32 + 255) / 256); if (blk > 8192) blk = 8192;
    zero_kernel<<<blk, 256, 0, stream>>>((uint32_t*)p, n32);
  };
  auto pack = [&](const float* s, int rows, int cols, bf16_t* dst, int Kp, int N) {
    size_t tot = (size_t)(Kp >> 5) * (N >> 4) * 512;
    pack_b_kernel<<<(int)((tot + 255) / 256), 256, 0, stream>>>(s, rows, cols, dst, Kp, N);
  };
  auto gemm = [&](const bf16_t* A, int lda, const bf16_t* Bp, float* C, int ldc,
                  int M, int N, int K, const float* bias, int act, int accum) {
    int tiles = (M >> 4) * (N >> 5);
    gemm_bf16_kernel<<<(tiles + 7) / 8, 256, 0, stream>>>(A, lda, Bp, C, ldc,
                                                          M, N, K, bias, act, accum);
  };

  // weights -> bf16 packed fragment order (zero padded)
  pack(l0fWih, 336, N4H, wih_l0f, KP_IN0, N4H);   pack(l0bWih, 336, N4H, wih_l0b, KP_IN0, N4H);
  pack(l0fWhh, NH_, N4H, whh_l0f, KP_H, N4H);     pack(l0bWhh, NH_, N4H, whh_l0b, KP_H, N4H);
  pack(l1fWih, 2 * NH_, N4H, wih_l1f, KP_2H, N4H);pack(l1bWih, 2 * NH_, N4H, wih_l1b, KP_2H, N4H);
  pack(l1fWhh, NH_, N4H, whh_l1f, KP_H, N4H);     pack(l1bWhh, NH_, N4H, whh_l1b, KP_H, N4H);
  pack(d0Wih, NE_, N4H, wih_d0, KP_E, N4H);       pack(d0Whh, NH_, N4H, whh_d0, KP_H, N4H);
  pack(d1Wih, NH_, N4H, wih_d1, KP_H, N4H);       pack(d1Whh, NH_, N4H, whh_d1, KP_H, N4H);
  pack(W_h, 2 * NH_, NH_, wh_bf, KP_2H, KP_H);
  pack(W_s, NH_, NH_, wsm_bf, KP_H, KP_H);
  pack(Vw, 3 * NH_, NV_, vw_bf, KP_3H, NV_);

  // init state / padded activation buffers
  zero(l0bf, (size_t)SB_ * KP_2H * 2);
  zero(eobf, (size_t)SB_ * KP_2H * 2);
  zero(h0f, NB_ * NH_ * 4); zero(c0f, NB_ * NH_ * 4);
  zero(h1f, NB_ * NH_ * 4); zero(c1f, NB_ * NH_ * 4);
  zero(h0bf, NB_ * KP_H * 2); zero(h1bf, NB_ * KP_H * 2);

  // ----- encoder ---------------------------------------------------------------------
  {
    size_t tot = (size_t)SB_ * KP_IN0;
    embed_concat_kernel<<<(int)((tot + 255) / 256), 256, 0, stream>>>(
        src, feat0, feat1, feat2, feat3, emb, fe0, fe1, fe2, fe3, xbf);
  }
  gemm(xbf, KP_IN0, wih_l0f, xiA, N4H, SB_, N4H, KP_IN0, nullptr, 0, 0);
  gemm(xbf, KP_IN0, wih_l0b, xiB, N4H, SB_, N4H, KP_IN0, nullptr, 0, 0);
  {
    ScanDir f{xiA, whh_l0f, l0fb, gbuf,             l0bf, nullptr, 0, 0};
    ScanDir b{xiB, whh_l0b, l0bb, gbuf + NB_ * N4H, l0bf, nullptr, 1, NH_};
    lstm_scan_kernel<<<2, 1024, 0, stream>>>(f, b, KP_2H);
  }
  gemm(l0bf, KP_2H, wih_l1f, xiA, N4H, SB_, N4H, KP_2H, nullptr, 0, 0);
  gemm(l0bf, KP_2H, wih_l1b, xiB, N4H, SB_, N4H, KP_2H, nullptr, 0, 0);
  {
    ScanDir f{xiA, whh_l1f, l1fb, gbuf,             eobf, eof, 0, 0};
    ScanDir b{xiB, whh_l1b, l1bb, gbuf + NB_ * N4H, eobf, eof, 1, NH_};
    lstm_scan_kernel<<<2, 1024, 0, stream>>>(f, b, KP_2H);
  }
  gemm(eobf, KP_2H, wh_bf, x1, KP_H, SB_, KP_H, KP_2H, nullptr, 0, 0);

  // ----- teacher-forced decoder ------------------------------------------------------
  for (int t = 0; t < NT_ - 1; ++t) {
    dec_embed_kernel<<<(NB_ * KP_E + 255) / 256, 256, 0, stream>>>(trg, t, emb, ebuf);
    gemm(ebuf, KP_E, wih_d0, gd, N4H, NB_, N4H, KP_E, nullptr, 0, 0);
    gemm(h0bf, KP_H, whh_d0, gd, N4H, NB_, N4H, KP_H, nullptr, 0, 1);
    dec_gate_kernel<<<(NB_ * NH_ + 255) / 256, 256, 0, stream>>>(gd, d0b, c0f, h0f, h0bf);
    gemm(h0bf, KP_H, wih_d1, gd, N4H, NB_, N4H, KP_H, nullptr, 0, 0);
    gemm(h1bf, KP_H, whh_d1, gd, N4H, NB_, N4H, KP_H, nullptr, 0, 1);
    dec_gate_kernel<<<(NB_ * NH_ + 255) / 256, 256, 0, stream>>>(gd, d1b, c1f, h1f, h1bf);
    gemm(h1bf, KP_H, wsm_bf, qbuf, KP_H, NB_, KP_H, KP_H, nullptr, 0, 0);
    attn_score_kernel<<<SB_ / 8, 256, 0, stream>>>(x1, qbuf, b_attn, v_vec, scores);
    softmax_kernel<<<NB_, 256, 0, stream>>>(scores, attn);
    context_kernel<<<(NB_ * 2 * NH_ + 255) / 256, 256, 0, stream>>>(attn, eof, ctx);
    build_gs_kernel<<<(NB_ * KP_3H + 255) / 256, 256, 0, stream>>>(h1f, ctx, gsbf);
    gemm(gsbf, KP_3H, vw_bf, pv, NV_, NB_, NV_, KP_3H, Vb, 1, 0);
    pz_kernel<<<1, NB_, 0, stream>>>(h1f, wz, bz, pz);
    {
      size_t tot = (size_t)NB_ * NV_;
      scale_store_kernel<<<(int)((tot + 255) / 256), 256, 0, stream>>>(pv, pz, out, t);
    }
    scatter_kernel<<<(NB_ * NS_ + 255) / 256, 256, 0, stream>>>(src, attn, pz, out, t);
  }
}
